// EfficientSparseCrossAttention_996432412694
// MI455X (gfx1250) — matrix operand order repaired and checked by reading, not compiled, not run on hardware
//
#include <hip/hip_runtime.h>
#include <hip/hip_bf16.h>

// ---------------------------------------------------------------------------
// Problem constants (from reference): bs=4, r=c=128, d=512, H=16, HD=32, MSH=32
// ---------------------------------------------------------------------------
#define BS   4
#define RC   128
#define DIM  512
#define NH   16
#define HD   32
#define MSH  32

typedef __attribute__((ext_vector_type(16))) __bf16 v16bf;
typedef __attribute__((ext_vector_type(8)))  float  v8f;

// ---- bf16 helpers (native hardware convert, RTNE) --------------------------
__device__ __forceinline__ __bf16 f2bf(float f) {
    return (__bf16)f;                       // hardware v_cvt, round-nearest-even
}
__device__ __forceinline__ unsigned short f2bfbits(float f) {
    union { __bf16 b; unsigned short u; } c;
    c.b = (__bf16)f;
    return c.u;
}
__device__ __forceinline__ __bf16 us2bf(unsigned short s) {
    union { unsigned short u; __bf16 b; } c; c.u = s; return c.b;
}
__device__ __forceinline__ v8f vzero() {
    v8f z = {0.f, 0.f, 0.f, 0.f, 0.f, 0.f, 0.f, 0.f};
    return z;
}
__device__ __forceinline__ v8f wmma_bf16(v16bf a, v16bf b, v8f c) {
    // v_wmma_f32_16x16x32_bf16 : D = A(16x32) * B(32x16) + C
    return __builtin_amdgcn_wmma_f32_16x16x32_bf16(
        /*neg_a=*/false, a, /*neg_b=*/false, b,
        /*c_mod=*/(short)0, c, /*reuse_a=*/false, /*reuse_b=*/false);
}

// ---- WMMA operand builders (CDNA5 wave32 VGPR layouts, ISA 7.12.2) ---------
// A (16x32, 16-bit): lane<16 holds row=lane, K in {0..7,16..23};
//                    lane>=16 holds row=lane-16, K in {8..15,24..31}.
template <typename G>
__device__ __forceinline__ v16bf load_a(int lane, G g) {
    v16bf a;
    int m  = lane & 15;
    int kb = (lane & 16) ? 8 : 0;
#pragma unroll
    for (int i = 0; i < 8; ++i) {
#pragma unroll
        for (int j = 0; j < 2; ++j) {
            int k = (i < 4) ? (kb + 2 * i + j) : (16 + kb + 2 * (i - 4) + j);
            a[2 * i + j] = g(m, k);
        }
    }
    return a;
}
// B (32x16, 16-bit): lane<16 holds N=lane, K in {0..7,16..23}; lane>=16 holds N=lane-16, K in {8..15,24..31}.
template <typename G>
__device__ __forceinline__ v16bf load_b(int lane, G g) {
    v16bf b;
    int n  = lane & 15;
    int kb = (lane & 16) ? 8 : 0;
#pragma unroll
    for (int e = 0; e < 16; ++e) b[e] = g((e < 8) ? (kb + e) : (16 + kb + (e - 8)), n);
    return b;
}
// C/D (16x16 f32): VGPR g, lane l -> (M = g + 8*(l>=16), N = l%16)

// ---------------------------------------------------------------------------
// Kernel 1: fully fused per-(b,h) pipeline.
//   grid = 64 (= BS*NH), block = 256 (8 waves). Wave w owns M-rows [16w,16w+16).
//   LDS overlay (54 KB):
//     [    0 .. 32K)  amat   bf16[128][128]   (phase>=4; first 16K doubles as Q,K)
//     [    0 ..  8K)  Q      bf16[128][32]    (phase 1-2)
//     [   8K .. 16K)  K      bf16[128][32]    (phase 1-2)
//     [  32K .. 40K)  V1     bf16[128][32]
//     [  40K .. 48K)  V2     bf16[128][32]
//     [  48K .. 52K)  red    f32 [8][128]     (col-softmax scratch; doubles as
//                                              B-stage bf16[32][64] in phase 1)
//     [  52K .. 53K)  wsm    f32 [162]        (per-head MLP weights)
// ---------------------------------------------------------------------------
__global__ __launch_bounds__(256) void fused_attn_kernel(
    const float* __restrict__ row_emb, const float* __restrict__ col_emb,
    const float* __restrict__ cost,    const float* __restrict__ Wqv1,
    const float* __restrict__ Wkv2,    const float* __restrict__ ms_W1,
    const float* __restrict__ ms_b1,   const float* __restrict__ ms_W2,
    const float* __restrict__ ms_b2,
    unsigned short* __restrict__ row_ws,   // bf16 [BS*RC][DIM]
    unsigned short* __restrict__ col_ws)   // bf16 [BS*RC][DIM]
{
    __shared__ __align__(16) unsigned char smem[54 * 1024];
    unsigned short* Qs   = (unsigned short*)(smem);
    unsigned short* Ks   = (unsigned short*)(smem + 8 * 1024);
    unsigned short* amat = (unsigned short*)(smem);
    unsigned short* V1s  = (unsigned short*)(smem + 32 * 1024);
    unsigned short* V2s  = (unsigned short*)(smem + 40 * 1024);
    float*          red  = (float*)(smem + 48 * 1024);
    unsigned short* stag = (unsigned short*)(smem + 48 * 1024);
    float*          wsm  = (float*)(smem + 52 * 1024);

    const int b    = blockIdx.x >> 4;
    const int h    = blockIdx.x & 15;
    const int tid  = threadIdx.x;
    const int wave = tid >> 5;
    const int lane = tid & 31;
    const int mb   = wave * 16;
    const float rscale = 0.17677669529663687f;  // 1/sqrt(HD)

    // ---------------- Phase 1: per-head projections via WMMA ----------------
    // Q,V1 = row_emb[b] @ Wqv1[:, {h, D+h} slices];  K,V2 = col_emb[b] @ Wkv2[...]
#pragma unroll 1
    for (int p = 0; p < 2; ++p) {
        const float* src = (p == 0 ? row_emb : col_emb) + (size_t)b * RC * DIM;
        const float* W   = (p == 0 ? Wqv1 : Wkv2);
        unsigned short* d0 = (p == 0 ? Qs : Ks);    // q / k   slice
        unsigned short* d1 = (p == 0 ? V1s : V2s);  // v1 / v2 slice
        v8f acc[4];
#pragma unroll
        for (int t = 0; t < 4; ++t) acc[t] = vzero();

#pragma unroll 1
        for (int kb = 0; kb < DIM; kb += 32) {
            __syncthreads();
            // Stage 32x64 slice of W (cols: 32 for q/k, 32 for v) as bf16.
            for (int e = tid; e < 32 * 64; e += 256) {
                int kk = e >> 6, cc = e & 63;
                int col = (cc < 32) ? (h * HD + cc) : (DIM + h * HD + (cc - 32));
                stag[e] = f2bfbits(W[(size_t)(kb + kk) * (2 * DIM) + col]);
            }
            __syncthreads();

            v16bf a = load_a(lane, [&](int m, int k) {
                return f2bf(src[(size_t)(mb + m) * DIM + kb + k]);
            });
#pragma unroll
            for (int t = 0; t < 4; ++t) {
                const int sel = t >> 1, nt = t & 1;
                v16bf bm = load_b(lane, [&](int k, int n) {
                    return us2bf(stag[k * 64 + sel * 32 + nt * 16 + n]);
                });
                acc[t] = wmma_bf16(a, bm, acc[t]);
            }
        }
        // Spill accumulators to LDS as bf16 [128][32].
#pragma unroll
        for (int t = 0; t < 4; ++t) {
            unsigned short* dst = (t < 2) ? d0 : d1;
            const int nb = (t & 1) * 16;
#pragma unroll
            for (int g = 0; g < 8; ++g) {
                int m = mb + g + ((lane & 16) ? 8 : 0);
                dst[m * HD + nb + (lane & 15)] = f2bfbits(acc[t][g]);
            }
        }
    }
    __syncthreads();

    // ---------------- Phase 2: logits = Q K^T  (K-dim = HD = 32) ------------
    v8f lg[8];   // 16x128 strip of logits per wave (later reused as l1)
    {
        v16bf a = load_a(lane, [&](int m, int k) {
            return us2bf(Qs[(mb + m) * HD + k]);
        });
#pragma unroll
        for (int nt = 0; nt < 8; ++nt) {
            v16bf bm = load_b(lane, [&](int k, int n) {
                return us2bf(Ks[(nt * 16 + n) * HD + k]);  // K^T: B[k][n] = K[n][k]
            });
            lg[nt] = wmma_bf16(a, bm, vzero());
        }
    }

    // ---------------- Phase 3: per-element 2->32->2 MLP + mask --------------
    if (tid < 32) {
        wsm[tid]       = ms_W1[h * 2 * MSH + tid];            // W1[h][0][:]
        wsm[32 + tid]  = ms_W1[h * 2 * MSH + MSH + tid];      // W1[h][1][:]
        wsm[64 + tid]  = ms_b1[h * MSH + tid];
        wsm[96 + tid]  = ms_W2[h * MSH * 2 + tid * 2 + 0];
        wsm[128 + tid] = ms_W2[h * MSH * 2 + tid * 2 + 1];
    }
    if (tid == 0) { wsm[160] = ms_b2[h * 2]; wsm[161] = ms_b2[h * 2 + 1]; }
    __syncthreads();   // also fences phase-2 reads of Q/K before amat overlay

    v8f l2[8];
    const float* costb = cost + (size_t)b * RC * RC;
    __builtin_prefetch(costb + (size_t)mb * RC, 0, 3);
#pragma unroll
    for (int nt = 0; nt < 8; ++nt) {
#pragma unroll
        for (int g = 0; g < 8; ++g) {
            int m = mb + g + ((lane & 16) ? 8 : 0);
            int n = nt * 16 + (lane & 15);
            float w   = costb[m * RC + n];
            float lgv = lg[nt][g] * rscale;
            float o0 = wsm[160], o1 = wsm[161];
#pragma unroll 4
            for (int mm = 0; mm < MSH; ++mm) {
                float hv = fmaxf(lgv * wsm[mm] + w * wsm[32 + mm] + wsm[64 + mm], 0.f);
                o0 += hv * wsm[96 + mm];
                o1 += hv * wsm[128 + mm];
            }
            bool msk = (w > 0.f);
            lg[nt][g] = msk ? o0 : -1e30f;   // l1
            l2[nt][g] = msk ? o1 : -1e30f;   // l2
        }
    }

    // ---------------- Phase 4a: row softmax of l1 (within wave) -------------
    // A row lives in one 16-lane half (fixed g, 8 n-tiles): shfl_xor 1..8.
#pragma unroll
    for (int g = 0; g < 8; ++g) {
        float mx = -3.0e38f;
#pragma unroll
        for (int nt = 0; nt < 8; ++nt) mx = fmaxf(mx, lg[nt][g]);
#pragma unroll
        for (int d = 1; d < 16; d <<= 1) mx = fmaxf(mx, __shfl_xor(mx, d, 32));
        float s = 0.f;
#pragma unroll
        for (int nt = 0; nt < 8; ++nt) { float z = __expf(lg[nt][g] - mx); lg[nt][g] = z; s += z; }
#pragma unroll
        for (int d = 1; d < 16; d <<= 1) s += __shfl_xor(s, d, 32);
        float inv = 1.f / ((s == 0.f) ? 1.f : s);
        int m = mb + g + ((lane & 16) ? 8 : 0);
#pragma unroll
        for (int nt = 0; nt < 8; ++nt)
            amat[m * RC + nt * 16 + (lane & 15)] = f2bfbits(lg[nt][g] * inv);
    }
    __syncthreads();

    // ---------------- Phase 5a: row_out = A1 @ V2 ---------------------------
    {
        v8f ro[2] = {vzero(), vzero()};
#pragma unroll 1
        for (int kb = 0; kb < RC; kb += 32) {
            v16bf a = load_a(lane, [&](int m, int k) {
                return us2bf(amat[(mb + m) * RC + kb + k]);
            });
#pragma unroll
            for (int t = 0; t < 2; ++t) {
                v16bf bm = load_b(lane, [&](int k, int n) {
                    return us2bf(V2s[(kb + k) * HD + t * 16 + n]);
                });
                ro[t] = wmma_bf16(a, bm, ro[t]);
            }
        }
#pragma unroll
        for (int t = 0; t < 2; ++t)
#pragma unroll
            for (int g = 0; g < 8; ++g) {
                int m = mb + g + ((lane & 16) ? 8 : 0);
                int n = t * 16 + (lane & 15);
                row_ws[(size_t)(b * RC + m) * DIM + h * HD + n] = f2bfbits(ro[t][g]);
            }
    }
    __syncthreads();   // everyone done reading A1 before rewriting amat

    // ---------------- Phase 4b: column softmax of l2 (cross-wave) -----------
    float cm[8];
#pragma unroll
    for (int nt = 0; nt < 8; ++nt) {
        float pm = -3.0e38f;
#pragma unroll
        for (int g = 0; g < 8; ++g) pm = fmaxf(pm, l2[nt][g]);
        pm = fmaxf(pm, __shfl_xor(pm, 16, 32));       // combine both 8-row halves
        if (lane < 16) red[wave * RC + nt * 16 + lane] = pm;
    }
    __syncthreads();
#pragma unroll
    for (int nt = 0; nt < 8; ++nt) {
        float m = -3.0e38f;
#pragma unroll
        for (int w = 0; w < 8; ++w) m = fmaxf(m, red[w * RC + nt * 16 + (lane & 15)]);
        cm[nt] = m;
    }
    __syncthreads();
#pragma unroll
    for (int nt = 0; nt < 8; ++nt) {
        float ps = 0.f;
#pragma unroll
        for (int g = 0; g < 8; ++g) { l2[nt][g] = __expf(l2[nt][g] - cm[nt]); ps += l2[nt][g]; }
        ps += __shfl_xor(ps, 16, 32);
        if (lane < 16) red[wave * RC + nt * 16 + lane] = ps;
    }
    __syncthreads();
#pragma unroll
    for (int nt = 0; nt < 8; ++nt) {
        float s = 0.f;
#pragma unroll
        for (int w = 0; w < 8; ++w) s += red[w * RC + nt * 16 + (lane & 15)];
        float inv = 1.f / ((s == 0.f) ? 1.f : s);
#pragma unroll
        for (int g = 0; g < 8; ++g) {
            int m = mb + g + ((lane & 16) ? 8 : 0);
            int n = nt * 16 + (lane & 15);
            amat[n * RC + m] = f2bfbits(l2[nt][g] * inv);   // store A2^T
        }
    }
    __syncthreads();

    // ---------------- Phase 5b: col_out = A2^T @ V1 -------------------------
    {
        v8f co[2] = {vzero(), vzero()};
#pragma unroll 1
        for (int kb = 0; kb < RC; kb += 32) {
            v16bf a = load_a(lane, [&](int m, int k) {
                return us2bf(amat[(mb + m) * RC + kb + k]);   // rows = c-index now
            });
#pragma unroll
            for (int t = 0; t < 2; ++t) {
                v16bf bm = load_b(lane, [&](int k, int n) {
                    return us2bf(V1s[(kb + k) * HD + t * 16 + n]);
                });
                co[t] = wmma_bf16(a, bm, co[t]);
            }
        }
#pragma unroll
        for (int t = 0; t < 2; ++t)
#pragma unroll
            for (int g = 0; g < 8; ++g) {
                int m = mb + g + ((lane & 16) ? 8 : 0);
                int n = t * 16 + (lane & 15);
                col_ws[(size_t)(b * RC + m) * DIM + h * HD + n] = f2bfbits(co[t][g]);
            }
    }
}

// ---------------------------------------------------------------------------
// Kernel 2: output projections. 32 blocks: sel(2) x 4x4 tiles of 128x128.
//   out[sel] (512x512) = (sel ? col_ws : row_ws)(bf16 512x512) @ Wo{1,2}
// ---------------------------------------------------------------------------
__global__ __launch_bounds__(256) void out_proj_kernel(
    const unsigned short* __restrict__ row_ws,
    const unsigned short* __restrict__ col_ws,
    const float* __restrict__ Wo1, const float* __restrict__ Wo2,
    float* __restrict__ out)
{
    __shared__ unsigned short stage[32 * 128];   // bf16 B tile per k-step (8KB)
    const int sel  = blockIdx.x >> 4;
    const int tile = blockIdx.x & 15;
    const int tm   = (tile >> 2) * 128;
    const int tn   = (tile & 3) * 128;
    const unsigned short* A = sel ? col_ws : row_ws;
    const float* W = sel ? Wo2 : Wo1;
    float* outp = out + (size_t)sel * (BS * RC * DIM);

    const int tid  = threadIdx.x;
    const int wave = tid >> 5;
    const int lane = tid & 31;
    const int mb   = tm + wave * 16;

    v8f acc[8];
#pragma unroll
    for (int nt = 0; nt < 8; ++nt) acc[nt] = vzero();

#pragma unroll 1
    for (int kb = 0; kb < DIM; kb += 32) {
        __syncthreads();
        for (int e = tid; e < 32 * 128; e += 256) {
            int kk = e >> 7, cc = e & 127;
            stage[e] = f2bfbits(W[(size_t)(kb + kk) * DIM + tn + cc]);
        }
        __syncthreads();
        v16bf a = load_a(lane, [&](int m, int k) {
            return us2bf(A[(size_t)(mb + m) * DIM + kb + k]);
        });
#pragma unroll
        for (int nt = 0; nt < 8; ++nt) {
            v16bf bm = load_b(lane, [&](int k, int n) {
                return us2bf(stage[k * 128 + nt * 16 + n]);
            });
            acc[nt] = wmma_bf16(a, bm, acc[nt]);
        }
    }
#pragma unroll
    for (int nt = 0; nt < 8; ++nt)
#pragma unroll
        for (int g = 0; g < 8; ++g) {
            int m = mb + g + ((lane & 16) ? 8 : 0);
            int n = tn + nt * 16 + (lane & 15);
            outp[(size_t)m * DIM + n] = acc[nt][g];
        }
}

// ---------------------------------------------------------------------------
extern "C" void kernel_launch(void* const* d_in, const int* in_sizes, int n_in,
                              void* d_out, int out_size, void* d_ws, size_t ws_size,
                              hipStream_t stream)
{
    (void)in_sizes; (void)n_in; (void)out_size; (void)ws_size;
    const float* row_emb = (const float*)d_in[0];
    const float* col_emb = (const float*)d_in[1];
    const float* cost    = (const float*)d_in[2];
    const float* Wqv1    = (const float*)d_in[3];
    const float* Wkv2    = (const float*)d_in[4];
    const float* ms_W1   = (const float*)d_in[5];
    const float* ms_b1   = (const float*)d_in[6];
    const float* ms_W2   = (const float*)d_in[7];
    const float* ms_b2   = (const float*)d_in[8];
    const float* Wo1     = (const float*)d_in[9];
    const float* Wo2     = (const float*)d_in[10];

    unsigned short* row_ws = (unsigned short*)d_ws;                 // bf16 512x512
    unsigned short* col_ws = row_ws + (size_t)BS * RC * DIM;        // bf16 512x512

    fused_attn_kernel<<<BS * NH, 256, 0, stream>>>(
        row_emb, col_emb, cost, Wqv1, Wkv2, ms_W1, ms_b1, ms_W2, ms_b2,
        row_ws, col_ws);

    out_proj_kernel<<<32, 256, 0, stream>>>(row_ws, col_ws, Wo1, Wo2,
                                            (float*)d_out);
}
